// SoftSOM_34574486733130
// MI455X (gfx1250) — matrix operand order repaired and checked by reading, not compiled, not run on hardware
//
#include <hip/hip_runtime.h>
#include <hip/hip_bf16.h>
#include <math.h>

// ---------------------------------------------------------------------------
// SoftSOM on MI455X (gfx1250, wave32)
//   - v_wmma_f32_16x16x32_f16 for the three N,K,D contractions (f32 accum)
//   - proto/grid/protoT pre-packed to f16 once (~1.7MB ws); GEMM hot loops
//     stream tiles with GLOBAL_LOAD_ASYNC_TO_LDS_B128, double-buffered so the
//     ASYNCcnt wait overlaps with WMMA on the other buffer
//   - norms / sqrt / exp / both normalizations in exact f32
// ---------------------------------------------------------------------------

#define NN   32768
#define KK   256
#define DD   1024
#define EPS_T 0.001f
#define EPS_W 1e-8f
#define PITCH 20   // LDS row pitch (uints): rows 16B-aligned, 20i mod 64 distinct -> conflict-free

typedef __attribute__((ext_vector_type(16))) _Float16 v16h;
typedef __attribute__((ext_vector_type(8)))  float    v8f;

union F16Vec { v16h h; unsigned u[8]; };

static __device__ inline unsigned pack2h(float a, float b) {
    union { _Float16 h[2]; unsigned u; } v;
    v.h[0] = (_Float16)a; v.h[1] = (_Float16)b;
    return v.u;
}

static __device__ inline float sigmoid_affine(float r) {
    return 1.0f / (1.0f + expf(-r)) * (1.0f - EPS_T) + EPS_T;
}

// LDS byte-offset of a shared pointer (flat LDS pointers carry the DS offset
// in their low 32 bits on AMDGPU)
static __device__ inline unsigned lds_off(const void* p) {
    return (unsigned)(size_t)p;
}

// async copy of 16 bytes: global (sbase + voff) -> LDS[dst]   (ASYNCcnt)
static __device__ inline void async_cp16(unsigned lds_dst, unsigned voff,
                                         const void* sbase) {
    asm volatile("global_load_async_to_lds_b128 %0, %1, %2 offset:0"
                 :: "v"(lds_dst), "v"(voff),
                    "s"((unsigned long long)(size_t)sbase)
                 : "memory");
}

static __device__ inline void async_wait0() {
    asm volatile("s_wait_asynccnt 0" ::: "memory");
}

// ---------------------------------------------------------------------------
// Kernel 0: one-time f32 -> f16 packing of prototypes / grid_pos (+ proto^T)
//   ph/gh : [256][512] uints, pair (2d,2d+1) along D
//   pt    : [1024][128] uints, pt[d][ku] = (proto[2ku][d], proto[2ku+1][d])
// ---------------------------------------------------------------------------
__global__ __launch_bounds__(256) void SoftSOM_cvt(
    const float* __restrict__ proto, const float* __restrict__ grid,
    unsigned* __restrict__ ph, unsigned* __restrict__ gh,
    unsigned* __restrict__ pt) {
    int idx = blockIdx.x * 256 + threadIdx.x;          // 0 .. 131071
    {
        int n = idx >> 9, du = idx & 511;
        const float* sp = proto + (size_t)n * DD + du * 2;
        const float* sg = grid  + (size_t)n * DD + du * 2;
        ph[idx] = pack2h(sp[0], sp[1]);
        gh[idx] = pack2h(sg[0], sg[1]);
    }
    {
        int d = idx >> 7, ku = idx & 127;
        float v0 = proto[(size_t)(2 * ku)     * DD + d];
        float v1 = proto[(size_t)(2 * ku + 1) * DD + d];
        pt[idx] = pack2h(v0, v1);
    }
}

// ---------------------------------------------------------------------------
// Kernel 1: exact f32 row squared-norms (wave per row, wave32 reduce)
// ---------------------------------------------------------------------------
__global__ __launch_bounds__(256) void SoftSOM_rownorm(const float* __restrict__ A,
                                                       float* __restrict__ out,
                                                       int rows) {
    int wave = (int)((blockIdx.x * blockDim.x + threadIdx.x) >> 5);
    int lane = threadIdx.x & 31;
    if (wave >= rows) return;
    const float* r = A + (size_t)wave * DD;
    float s = 0.0f;
    #pragma unroll 4
    for (int i = lane; i < DD; i += 32) { float v = r[i]; s += v * v; }
    #pragma unroll
    for (int off = 16; off > 0; off >>= 1) s += __shfl_xor(s, off, 32);
    if (lane == 0) out[wave] = s;
}

// ---------------------------------------------------------------------------
// Kernel 2: gate = softmax(gate_logits / gate_temp)
// ---------------------------------------------------------------------------
__global__ __launch_bounds__(256) void SoftSOM_gate(const float* __restrict__ logits,
                                                    const float* __restrict__ gt_raw,
                                                    float* __restrict__ gate) {
    __shared__ float sm[256];
    int t = threadIdx.x;
    float gtemp = sigmoid_affine(gt_raw[0]);
    float v = logits[t] / gtemp;
    sm[t] = v; __syncthreads();
    for (int s = 128; s > 0; s >>= 1) { if (t < s) sm[t] = fmaxf(sm[t], sm[t + s]); __syncthreads(); }
    float m = sm[0]; __syncthreads();
    float e = expf(v - m);
    sm[t] = e; __syncthreads();
    for (int s = 128; s > 0; s >>= 1) { if (t < s) sm[t] += sm[t + s]; __syncthreads(); }
    gate[t] = e / sm[0];
}

// ---------------------------------------------------------------------------
// Kernel 3: fused dual-cdist + exp + double normalization + gate  -> w [N,K]
//   Block: 256 threads (8 waves), 32 rows of x, all K=256 columns.
//   Double-buffered LDS tiles; proto/grid via async DMA, x via VALU cvt.
// ---------------------------------------------------------------------------
struct Tiles { unsigned x[32][PITCH]; unsigned p[256][PITCH]; unsigned g[256][PITCH]; };

union __align__(16) SmemA {
    Tiles t[2];                 // ~87 KB
    float w[32][264];           // f32 w tile for the normalization phase
};

static __device__ inline void distw_load(Tiles* T, const float* __restrict__ x,
                                         size_t row0, int d0,
                                         const unsigned* ph, const unsigned* gh,
                                         int tid) {
    // proto/grid tiles: 256 rows x 64B each -> 1024 b128 DMAs per tile
    #pragma unroll
    for (int it = 0; it < 4; ++it) {
        int i = tid + it * 256;
        int n = i >> 2, g4 = (i & 3) * 4;
        unsigned voff = (unsigned)(n * 2048 + d0 * 2 + g4 * 4);
        async_cp16(lds_off(&T->p[n][g4]), voff, ph);
        async_cp16(lds_off(&T->g[n][g4]), voff, gh);
    }
    // x tile: 32 rows x 32 f32 -> f16 (x is read only once per block)
    #pragma unroll
    for (int it = 0; it < 2; ++it) {
        int i = tid + it * 256;
        int m = i >> 4, ku = i & 15;
        const float* s = x + (row0 + m) * DD + d0 + ku * 2;
        T->x[m][ku] = pack2h(s[0], s[1]);
    }
}

__global__ __launch_bounds__(256) void SoftSOM_distw(
    const float* __restrict__ x,
    const unsigned* __restrict__ ph, const unsigned* __restrict__ gh,
    const float* __restrict__ temp_raw,
    const float* __restrict__ xn, const float* __restrict__ pn,
    const float* __restrict__ gn, const float* __restrict__ gate,
    float* __restrict__ w_out) {

    __shared__ SmemA sm;

    const int tid   = threadIdx.x;
    const int lane  = tid & 31;
    const int wid   = tid >> 5;
    const int halfq = lane >> 4;
    const int l16   = lane & 15;
    const int mSub  = wid >> 2;      // 0..1
    const int nBase = (wid & 3) * 64;
    const size_t row0 = (size_t)blockIdx.x * 32;

    v8f accP[4] = {}, accG[4] = {};

    distw_load(&sm.t[0], x, row0, 0, ph, gh, tid);
    async_wait0();
    __syncthreads();

    for (int it = 0; it < DD / 32; ++it) {
        Tiles* cur = &sm.t[it & 1];
        if (it + 1 < DD / 32)       // prefetch next chunk into the other buffer
            distw_load(&sm.t[(it + 1) & 1], x, row0, (it + 1) * 32, ph, gh, tid);

        // A fragment (16x32 f16): row M=l16 per lane; K packed 2/VGPR per ISA layout
        F16Vec a;
        const unsigned* xr = cur->x[mSub * 16 + l16];
        #pragma unroll
        for (int i = 0; i < 4; ++i) a.u[i]     = xr[halfq * 4 + i];
        #pragma unroll
        for (int i = 0; i < 4; ++i) a.u[4 + i] = xr[8 + halfq * 4 + i];

        #pragma unroll
        for (int ns = 0; ns < 4; ++ns) {
            int n = nBase + ns * 16 + l16;       // B column per lane
            F16Vec bp, bg;
            const unsigned* pr = cur->p[n];
            const unsigned* gr = cur->g[n];
            #pragma unroll
            for (int j = 0; j < 8; ++j) { bp.u[j] = pr[(j >> 2) * 8 + halfq * 4 + (j & 3)]; bg.u[j] = gr[(j >> 2) * 8 + halfq * 4 + (j & 3)]; }
            accP[ns] = __builtin_amdgcn_wmma_f32_16x16x32_f16(
                false, a.h, false, bp.h, (short)0, accP[ns], false, false);
            accG[ns] = __builtin_amdgcn_wmma_f32_16x16x32_f16(
                false, a.h, false, bg.h, (short)0, accG[ns], false, false);
        }
        async_wait0();     // next buffer DMA complete (overlapped with the WMMAs above)
        __syncthreads();   // everyone done reading cur; safe to overwrite next iter
    }

    const float temp = sigmoid_affine(temp_raw[0]);
    const float invT = 1.0f / temp;

    // union region repurposed for the f32 w tile (tiles dead after last barrier)
    #pragma unroll
    for (int ns = 0; ns < 4; ++ns) {
        int n = nBase + ns * 16 + l16;
        float pnv = pn[n], gnv = gn[n];
        #pragma unroll
        for (int r = 0; r < 8; ++r) {
            int m = mSub * 16 + r + 8 * halfq;   // C layout: VGPR r -> M = r + 8*half
            float xv = xn[row0 + m];
            float s1 = fmaxf(xv + pnv - 2.0f * accP[ns][r], 1e-12f);
            float s2 = fmaxf(xv + gnv - 2.0f * accG[ns][r], 1e-12f);
            float dt = sqrtf(s1) + sqrtf(s2);
            sm.w[m][n] = expf(-dt * invT);
        }
    }
    __syncthreads();

    // Row-wise: w/=sum+eps; w*=gate; w/=sum+eps  (8 threads per row, 32 cols each)
    {
        int rrow = tid >> 3;
        int sub  = tid & 7;
        float* wr = sm.w[rrow];
        float s = 0.0f;
        #pragma unroll 8
        for (int c = sub * 32; c < sub * 32 + 32; ++c) s += wr[c];
        #pragma unroll
        for (int off = 4; off > 0; off >>= 1) s += __shfl_xor(s, off, 8);
        float inv1 = 1.0f / (s + EPS_W);
        float s2 = 0.0f;
        #pragma unroll 8
        for (int c = sub * 32; c < sub * 32 + 32; ++c) {
            float u = wr[c] * inv1 * gate[c];
            wr[c] = u; s2 += u;
        }
        #pragma unroll
        for (int off = 4; off > 0; off >>= 1) s2 += __shfl_xor(s2, off, 8);
        float inv2 = 1.0f / (s2 + EPS_W);
        float* dst = w_out + (row0 + rrow) * KK;
        #pragma unroll 8
        for (int c = sub * 32; c < sub * 32 + 32; ++c) dst[c] = wr[c] * inv2;
    }
}

// ---------------------------------------------------------------------------
// Kernel 4: blended = w @ prototypes   via pre-transposed f16 proto^T
//   Block: 256 threads, 32 rows x 128 cols, double-buffered. grid=(D/128,N/32)
// ---------------------------------------------------------------------------
struct BTiles { unsigned w[32][PITCH]; unsigned p[128][PITCH]; };

static __device__ inline void blend_load(BTiles* T, const float* __restrict__ w,
                                         size_t row0, int k0, int n0,
                                         const unsigned* pt, int tid) {
    // proto^T tile: 128 rows (d) x 64B of K -> 512 b128 DMAs
    #pragma unroll
    for (int it = 0; it < 2; ++it) {
        int i = tid + it * 256;
        int n = i >> 2, g4 = (i & 3) * 4;
        unsigned voff = (unsigned)((n0 + n) * 512 + k0 * 2 + g4 * 4);
        async_cp16(lds_off(&T->p[n][g4]), voff, pt);
    }
    // w tile: 32 rows x 32 f32 -> f16
    #pragma unroll
    for (int it = 0; it < 2; ++it) {
        int i = tid + it * 256;
        int m = i >> 4, ku = i & 15;
        const float* s = w + (row0 + m) * KK + k0 + ku * 2;
        T->w[m][ku] = pack2h(s[0], s[1]);
    }
}

__global__ __launch_bounds__(256) void SoftSOM_blend(
    const float* __restrict__ w, const unsigned* __restrict__ pt,
    float* __restrict__ out) {

    __shared__ __align__(16) BTiles bt[2];

    const int tid   = threadIdx.x;
    const int lane  = tid & 31;
    const int wid   = tid >> 5;
    const int halfq = lane >> 4;
    const int l16   = lane & 15;
    const int mSub  = wid >> 2;
    const int nSubB = (wid & 3) * 2;
    const size_t row0 = (size_t)blockIdx.y * 32;
    const int n0 = blockIdx.x * 128;

    v8f acc[2] = {};

    blend_load(&bt[0], w, row0, 0, n0, pt, tid);
    async_wait0();
    __syncthreads();

    for (int it = 0; it < KK / 32; ++it) {
        BTiles* cur = &bt[it & 1];
        if (it + 1 < KK / 32)
            blend_load(&bt[(it + 1) & 1], w, row0, (it + 1) * 32, n0, pt, tid);

        F16Vec a;
        const unsigned* wr = cur->w[mSub * 16 + l16];
        #pragma unroll
        for (int i = 0; i < 4; ++i) a.u[i]     = wr[halfq * 4 + i];
        #pragma unroll
        for (int i = 0; i < 4; ++i) a.u[4 + i] = wr[8 + halfq * 4 + i];

        #pragma unroll
        for (int ns = 0; ns < 2; ++ns) {
            int n = (nSubB + ns) * 16 + l16;
            F16Vec b;
            const unsigned* pr = cur->p[n];
            #pragma unroll
            for (int j = 0; j < 8; ++j) b.u[j] = pr[(j >> 2) * 8 + halfq * 4 + (j & 3)];
            acc[ns] = __builtin_amdgcn_wmma_f32_16x16x32_f16(
                false, a.h, false, b.h, (short)0, acc[ns], false, false);
        }
        async_wait0();
        __syncthreads();
    }

    #pragma unroll
    for (int ns = 0; ns < 2; ++ns) {
        int n = n0 + (nSubB + ns) * 16 + l16;
        #pragma unroll
        for (int r = 0; r < 8; ++r) {
            int m = mSub * 16 + r + 8 * halfq;
            out[(row0 + m) * DD + n] = acc[ns][r];
        }
    }
}

// ---------------------------------------------------------------------------
// Launch
// ---------------------------------------------------------------------------
extern "C" void kernel_launch(void* const* d_in, const int* in_sizes, int n_in,
                              void* d_out, int out_size, void* d_ws, size_t ws_size,
                              hipStream_t stream) {
    const float* x         = (const float*)d_in[0];
    const float* proto     = (const float*)d_in[1];
    const float* grid_pos  = (const float*)d_in[2];
    const float* temp_raw  = (const float*)d_in[3];
    const float* gtemp_raw = (const float*)d_in[4];
    const float* glogits   = (const float*)d_in[5];

    float* blended = (float*)d_out;                      // [N, D]
    float* w_out   = blended + (size_t)NN * DD;          // [N, K]

    // workspace layout: norms/gate then f16 packs (total < 2 MB)
    char* ws = (char*)d_ws;
    float*    xn   = (float*)ws;                         // [N]
    float*    pn   = xn + NN;                            // [K]
    float*    gn   = pn + KK;                            // [K]
    float*    gate = gn + KK;                            // [K]
    unsigned* ph   = (unsigned*)(ws + (1u << 18));       // [256][512] u  (512 KB)
    unsigned* gh   = ph + KK * (DD / 2);                 // [256][512] u  (512 KB)
    unsigned* pt   = gh + KK * (DD / 2);                 // [1024][128] u (512 KB)

    SoftSOM_cvt<<<(KK * DD / 2) / 256, 256, 0, stream>>>(proto, grid_pos, ph, gh, pt);
    SoftSOM_rownorm<<<NN / 8, 256, 0, stream>>>(x, xn, NN);
    SoftSOM_rownorm<<<KK / 8, 256, 0, stream>>>(proto, pn, KK);
    SoftSOM_rownorm<<<KK / 8, 256, 0, stream>>>(grid_pos, gn, KK);
    SoftSOM_gate<<<1, 256, 0, stream>>>(glogits, gtemp_raw, gate);

    SoftSOM_distw<<<NN / 32, 256, 0, stream>>>(x, ph, gh, temp_raw,
                                               xn, pn, gn, gate, w_out);

    SoftSOM_blend<<<dim3(DD / 128, NN / 32), 256, 0, stream>>>(w_out, pt, blended);
}
